// BayesianLinear_4930622456479
// MI455X (gfx1250) — matrix operand order repaired and checked by reading, not compiled, not run on hardware
//
#include <hip/hip_runtime.h>

// ---------------------------------------------------------------------------
// BayesianLinear: y = x @ (mu + softplus(rho)*eps)^T + (bmu + softplus(brho)*beps)
// x:[8192,4096] f32 -> y:[8192,4096] f32
//
// MI455X (gfx1250) plan:
//   * prep: sample W / x into bf16 hi/lo pairs (split-precision bf16x3)
//   * GEMM: TDM (tensor_load_to_lds) double-buffered LDS staging,
//           ds_load_b128 fragment reads (conflict-free via TDM LDS padding),
//           v_wmma_f32_16x16x32_bf16 with f32 accumulators:
//              acc += Ah*Bh + Ah*Bl + Al*Bh
//   * block tile 128x128, 8 waves (4M x 2N), wave tile 32x64, K-slab 64.
//     LDS: 2 buffers x 4 tiles x (128 rows x 72 bf16) = 144 KB -> 2 WG/WGP.
// ---------------------------------------------------------------------------

typedef __bf16 bf16;
typedef __attribute__((ext_vector_type(16))) __bf16 v16bf;
typedef __attribute__((ext_vector_type(8)))  __bf16 v8bf;
typedef __attribute__((ext_vector_type(4)))  __bf16 v4bf;
typedef __attribute__((ext_vector_type(8)))  float  v8f;
typedef __attribute__((ext_vector_type(4)))  unsigned int uint32x4;
typedef __attribute__((ext_vector_type(8)))  int          int32x8;
typedef __attribute__((ext_vector_type(4)))  int          int32x4;

#define IN_F   4096
#define OUT_F  4096
#define TOKENS 8192

#define KSLAB    64            // K elements staged per pipeline stage
#define LDS_PITCH 72           // 64 elems + 8 elem pad (TDM pad: 4 DW per 32 DW)
#define TILE_ELEMS (128 * LDS_PITCH)

__device__ __forceinline__ float softplus_f(float r) {
  return fmaxf(r, 0.0f) + log1pf(expf(-fabsf(r)));
}

// ---------------- prep kernels --------------------------------------------

__global__ __launch_bounds__(256) void prep_w_kernel(
    const float4* __restrict__ mu, const float4* __restrict__ rho,
    const float4* __restrict__ eps, v4bf* __restrict__ hi,
    v4bf* __restrict__ lo, int n4) {
  int i = blockIdx.x * 256 + threadIdx.x;
  if (i >= n4) return;
  float4 m = mu[i], r = rho[i], e = eps[i];
  float w[4];
  w[0] = m.x + softplus_f(r.x) * e.x;
  w[1] = m.y + softplus_f(r.y) * e.y;
  w[2] = m.z + softplus_f(r.z) * e.z;
  w[3] = m.w + softplus_f(r.w) * e.w;
  v4bf h, l;
#pragma unroll
  for (int j = 0; j < 4; ++j) {
    bf16 hh = (bf16)w[j];
    h[j] = hh;
    l[j] = (bf16)(w[j] - (float)hh);
  }
  hi[i] = h;
  lo[i] = l;
}

__global__ __launch_bounds__(256) void prep_x_kernel(
    const float4* __restrict__ x, v4bf* __restrict__ hi,
    v4bf* __restrict__ lo, int n4) {
  int i = blockIdx.x * 256 + threadIdx.x;
  if (i >= n4) return;
  float4 v = x[i];
  float w[4] = {v.x, v.y, v.z, v.w};
  v4bf h, l;
#pragma unroll
  for (int j = 0; j < 4; ++j) {
    bf16 hh = (bf16)w[j];
    h[j] = hh;
    l[j] = (bf16)(w[j] - (float)hh);
  }
  hi[i] = h;
  lo[i] = l;
}

__global__ __launch_bounds__(256) void prep_b_kernel(
    const float* __restrict__ mu, const float* __restrict__ rho,
    const float* __restrict__ eps, float* __restrict__ b, int n) {
  int i = blockIdx.x * 256 + threadIdx.x;
  if (i < n) b[i] = mu[i] + softplus_f(rho[i]) * eps[i];
}

// ---------------- TDM descriptor + issue -----------------------------------

// Issue one TDM 2D tile load: 128 rows x 64 bf16, row stride 4096 elems in
// memory, LDS rows padded 128B -> 144B (pad 4 DW after every 32 DW).
// D# layout per CDNA5 ISA ch.8 (group0: 4 dwords, group1: 8 dwords).
__device__ __forceinline__ void tdm_load_tile(unsigned int lds_addr,
                                              unsigned long long gaddr,
                                              unsigned int tensor_rows) {
  uint32x4 g0;
  g0[0] = 1u;                                   // count=1, no gather
  g0[1] = lds_addr;                             // lds_addr [63:32]
  g0[2] = (unsigned int)gaddr;                  // global_addr low [95:64]
  g0[3] = (unsigned int)(gaddr >> 32)           // global_addr hi  [120:96]
          | (2u << 30);                         // type=2 ("image") [127:126]

  int32x8 g1;
  g1[0] = (int)((1u << 16)                      // data_size = 2 bytes
                | (1u << 20)                    // pad_enable
                | (4u << 22)                    // pad_interval: 32 DWORDs
                | (3u << 25));                  // pad_amount:   4 DWORDs
  g1[1] = (int)(((unsigned)IN_F & 0xFFFFu) << 16);       // tensor_dim0 lo16
  g1[2] = (int)((tensor_rows & 0xFFFFu) << 16);          // tensor_dim1 lo16
  g1[3] = (int)((unsigned)KSLAB << 16);                  // tile_dim0 = 64
  g1[4] = (int)128u;                                     // tile_dim1 = 128 rows
  g1[5] = (int)(unsigned)IN_F;                           // tensor_dim0_stride
  g1[6] = 0;
  g1[7] = 0;

  int32x4 z4 = {0, 0, 0, 0};
#if defined(__clang_major__) && (__clang_major__ >= 23)
  int32x8 z8 = {0, 0, 0, 0, 0, 0, 0, 0};
  __builtin_amdgcn_tensor_load_to_lds(g0, g1, z4, z4, z8, 0);
#else
  __builtin_amdgcn_tensor_load_to_lds(g0, g1, z4, z4, 0);
#endif
}

__device__ __forceinline__ unsigned int lds_addr_of(const bf16* p) {
  return (unsigned int)(unsigned long long)(const void*)p;
}

// ---------------- GEMM ----------------------------------------------------

__device__ __forceinline__ v16bf cat8(v8bf a, v8bf b) {
  return __builtin_shufflevector(a, b, 0, 1, 2, 3, 4, 5, 6, 7,
                                       8, 9, 10, 11, 12, 13, 14, 15);
}

__global__ __launch_bounds__(256) void bayes_gemm_tdm(
    const bf16* __restrict__ xh, const bf16* __restrict__ xl,
    const bf16* __restrict__ wh, const bf16* __restrict__ wl,
    const float* __restrict__ bias, float* __restrict__ out) {
  extern __shared__ __attribute__((aligned(16))) bf16 smem[];
  // smem layout: [buf 0..1][tile: 0=Ah 1=Al 2=Bh 3=Bl][128 rows * 72 elems]

  const int lane = threadIdx.x & 31;
  const int wave = threadIdx.x >> 5;
  const int half = lane >> 4;
  const int l16  = lane & 15;

  const int blockM = blockIdx.y * 128;
  const int blockN = blockIdx.x * 128;

  // global byte addresses of this block's K-slab streams
  const unsigned long long aH0 =
      (unsigned long long)(const void*)xh + (unsigned long long)blockM * IN_F * 2;
  const unsigned long long aL0 =
      (unsigned long long)(const void*)xl + (unsigned long long)blockM * IN_F * 2;
  const unsigned long long bH0 =
      (unsigned long long)(const void*)wh + (unsigned long long)blockN * IN_F * 2;
  const unsigned long long bL0 =
      (unsigned long long)(const void*)wl + (unsigned long long)blockN * IN_F * 2;

  // per-lane LDS fragment element offsets (WMMA register layouts):
  //   A (16x32, 16-bit): lane = row l16; K chunks [8*half,+8) and +16
  //   B (32x16, 16-bit): lane = col l16; K chunks [8*half,+8) and +16
  const int aOff = ((wave & 3) * 32 + l16) * LDS_PITCH + 8 * half;
  const int bOff = ((wave >> 2) * 64 + l16) * LDS_PITCH + 8 * half;

  v8f acc[2][4];
  const v8f vz = {0.f, 0.f, 0.f, 0.f, 0.f, 0.f, 0.f, 0.f};
#pragma unroll
  for (int mt = 0; mt < 2; ++mt)
#pragma unroll
    for (int nt = 0; nt < 4; ++nt) acc[mt][nt] = vz;

  auto issue_slab = [&](int buf, int k0) {
    const unsigned long long koff = (unsigned long long)k0 * 2;
    bf16* base = smem + (size_t)buf * 4 * TILE_ELEMS;
    tdm_load_tile(lds_addr_of(base + 0 * TILE_ELEMS), aH0 + koff, TOKENS);
    tdm_load_tile(lds_addr_of(base + 1 * TILE_ELEMS), aL0 + koff, TOKENS);
    tdm_load_tile(lds_addr_of(base + 2 * TILE_ELEMS), bH0 + koff, OUT_F);
    tdm_load_tile(lds_addr_of(base + 3 * TILE_ELEMS), bL0 + koff, OUT_F);
  };

  if (wave == 0) issue_slab(0, 0);  // prologue: fill buffer 0

  const int NK = IN_F / KSLAB;  // 64 pipeline stages
  for (int kt = 0; kt < NK; ++kt) {
    const int cur = kt & 1;

    // current slab's 4 TDM transfers are the only outstanding ones
    __builtin_amdgcn_s_wait_tensorcnt(0);
    __syncthreads();  // publish DMA'd LDS to all waves

    // kick next slab's DMA; overlaps with the 48 WMMAs below.
    // (buf cur^1 readers all finished at the previous end-barrier)
    if (wave == 0 && kt + 1 < NK) issue_slab(cur ^ 1, (kt + 1) * KSLAB);

    const bf16* Ah = smem + ((size_t)cur * 4 + 0) * TILE_ELEMS;
    const bf16* Al = smem + ((size_t)cur * 4 + 1) * TILE_ELEMS;
    const bf16* Bh = smem + ((size_t)cur * 4 + 2) * TILE_ELEMS;
    const bf16* Bl = smem + ((size_t)cur * 4 + 3) * TILE_ELEMS;

#pragma unroll
    for (int ks = 0; ks < 2; ++ks) {  // two k=32 sub-steps per slab
      const int k = ks * 32;
      v16bf ah[2], al[2], bh[4], bl[4];
#pragma unroll
      for (int mt = 0; mt < 2; ++mt) {
        const int o = aOff + mt * (16 * LDS_PITCH) + k;
        ah[mt] = cat8(*(const v8bf*)(Ah + o), *(const v8bf*)(Ah + o + 16));
        al[mt] = cat8(*(const v8bf*)(Al + o), *(const v8bf*)(Al + o + 16));
      }
#pragma unroll
      for (int nt = 0; nt < 4; ++nt) {
        const int o = bOff + nt * (16 * LDS_PITCH) + k;
        bh[nt] = cat8(*(const v8bf*)(Bh + o), *(const v8bf*)(Bh + o + 16));
        bl[nt] = cat8(*(const v8bf*)(Bl + o), *(const v8bf*)(Bl + o + 16));
      }
#pragma unroll
      for (int mt = 0; mt < 2; ++mt)
#pragma unroll
        for (int nt = 0; nt < 4; ++nt) {
          acc[mt][nt] = __builtin_amdgcn_wmma_f32_16x16x32_bf16(
              false, ah[mt], false, bh[nt], (short)0, acc[mt][nt], false, false);
          acc[mt][nt] = __builtin_amdgcn_wmma_f32_16x16x32_bf16(
              false, ah[mt], false, bl[nt], (short)0, acc[mt][nt], false, false);
          acc[mt][nt] = __builtin_amdgcn_wmma_f32_16x16x32_bf16(
              false, al[mt], false, bh[nt], (short)0, acc[mt][nt], false, false);
        }
    }
    __syncthreads();  // all waves done reading buf[cur] before its refill
  }

  // store: C/D layout -> VGPR v holds (m = v + 8*half, n = l16) of each tile
  const int waveM = blockM + (wave & 3) * 32;
  const int waveN = blockN + (wave >> 2) * 64;
#pragma unroll
  for (int mt = 0; mt < 2; ++mt) {
#pragma unroll
    for (int nt = 0; nt < 4; ++nt) {
      int col = waveN + 16 * nt + l16;
      float bv = bias[col];
      float* orow = out + (size_t)(waveM + 16 * mt + 8 * half) * OUT_F + col;
#pragma unroll
      for (int v = 0; v < 8; ++v)
        orow[(size_t)v * OUT_F] = acc[mt][nt][v] + bv;
    }
  }
}

// ---------------- launch ---------------------------------------------------

extern "C" void kernel_launch(void* const* d_in, const int* in_sizes, int n_in,
                              void* d_out, int out_size, void* d_ws,
                              size_t ws_size, hipStream_t stream) {
  (void)in_sizes; (void)n_in; (void)out_size; (void)ws_size;

  const float* x    = (const float*)d_in[0];
  const float* wmu  = (const float*)d_in[1];
  const float* wrho = (const float*)d_in[2];
  const float* bmu  = (const float*)d_in[3];
  const float* brho = (const float*)d_in[4];
  const float* epsw = (const float*)d_in[5];
  const float* epsb = (const float*)d_in[6];
  float* out = (float*)d_out;

  const size_t wElems = (size_t)OUT_F * IN_F;
  const size_t xElems = (size_t)TOKENS * IN_F;

  char* ws = (char*)d_ws;
  bf16* w_hi = (bf16*)ws;
  bf16* w_lo = w_hi + wElems;
  bf16* x_hi = w_lo + wElems;
  bf16* x_lo = x_hi + xElems;
  float* bias = (float*)(x_lo + xElems);

  {
    int n4 = (int)(wElems / 4);
    prep_w_kernel<<<n4 / 256, 256, 0, stream>>>(
        (const float4*)wmu, (const float4*)wrho, (const float4*)epsw,
        (v4bf*)w_hi, (v4bf*)w_lo, n4);
  }
  {
    int n4 = (int)(xElems / 4);
    prep_x_kernel<<<n4 / 256, 256, 0, stream>>>(
        (const float4*)x, (v4bf*)x_hi, (v4bf*)x_lo, n4);
  }
  prep_b_kernel<<<(OUT_F + 255) / 256, 256, 0, stream>>>(bmu, brho, epsb,
                                                         bias, OUT_F);
  {
    dim3 grid(OUT_F / 128, TOKENS / 128);
    const size_t smem_bytes =
        (size_t)2 * 4 * TILE_ELEMS * sizeof(bf16);  // 147456 B
    bayes_gemm_tdm<<<grid, 256, smem_bytes, stream>>>(x_hi, x_lo, w_hi, w_lo,
                                                      bias, out);
  }
}
